// PNA_87205015978669
// MI455X (gfx1250) — matrix operand order repaired and checked by reading, not compiled, not run on hardware
//
#include <hip/hip_runtime.h>
#include <hip/hip_bf16.h>
#include <math.h>

// ---------------------------------------------------------------------------
// PNAConv forward for MI455X (gfx1250, wave32, WMMA 16x16x32 bf16 -> f32 acc)
// ---------------------------------------------------------------------------

typedef __attribute__((ext_vector_type(16))) __bf16 v16bf;
typedef __attribute__((ext_vector_type(8)))  float  v8f;

union FragU { unsigned u[8]; v16bf v; };   // 16 bf16 == 8 dwords

__device__ __forceinline__ unsigned pack_bf2(float a, float b) {
    // round-to-nearest-even f32 -> bf16, pack two into one dword
    unsigned ua = __float_as_uint(a); ua = ua + 0x7fffu + ((ua >> 16) & 1u);
    unsigned ub = __float_as_uint(b); ub = ub + 0x7fffu + ((ub >> 16) & 1u);
    return (ua >> 16) | (ub & 0xffff0000u);
}

// order-preserving float<->uint for atomicMin/atomicMax
__device__ __forceinline__ unsigned fenc(float f) {
    unsigned u = __float_as_uint(f);
    return (u & 0x80000000u) ? ~u : (u | 0x80000000u);
}
__device__ __forceinline__ float fdec(unsigned u) {
    unsigned b = (u & 0x80000000u) ? (u & 0x7fffffffu) : ~u;
    return __uint_as_float(b);
}

// ---------------------------------------------------------------------------
// Generic GEMM: C[M,128] = A[M,K] * B[K,128] (+bias) (opt ReLU)
// A is f32 (ABF16=0) or bf16 (ABF16=1), row-major with leading dim lda.
// Block: 256 thr = 8 waves; tile 128(M)x128(N); wave = 32x64; 2x4 WMMA accs.
// Fast path (full M-tile, full K-step): vector loads, no guards, prefetch.
// ---------------------------------------------------------------------------
#define GSTR 17   // dword stride per LDS row: 16 K-pairs + 1 pad

template <int ABF16>
__global__ __launch_bounds__(256)
void gemm_wmma(const void* __restrict__ Aptr, int lda,
               const float* __restrict__ B, const float* __restrict__ bias,
               float* __restrict__ C, int M, int K, int doRelu)
{
    __shared__ unsigned AsU[128 * GSTR];   // As[row][kpair] bf16x2
    __shared__ unsigned BsU[128 * GSTR];   // Bs[col][kpair] bf16x2 (K-contig)

    const int tid  = threadIdx.x;
    const int lane = tid & 31;
    const int wave = tid >> 5;
    const int wm   = wave & 3;     // 32-row strip within tile
    const int wn   = wave >> 2;    // 64-col strip within tile
    const int blockRow = blockIdx.x * 128;
    const bool fullM = (blockRow + 128) <= M;

    v8f acc[2][4];
    #pragma unroll
    for (int i = 0; i < 2; ++i)
        #pragma unroll
        for (int j = 0; j < 4; ++j)
            #pragma unroll
            for (int e = 0; e < 8; ++e) acc[i][j][e] = 0.0f;

    const float*    Af  = (const float*)Aptr;
    const unsigned* Abf = (const unsigned*)Aptr;  // bf16 pairs (lda, K even)

    for (int k0 = 0; k0 < K; k0 += 32) {
        if (fullM && (k0 + 32) <= K) {
            // ---------------- fast path: no guards, vector loads ----------
            // A tile: 128 rows x 8 quads (4 floats / 2 dwords each)
            #pragma unroll
            for (int it = 0; it < 4; ++it) {
                int t  = tid + it * 256;           // 0..1023
                int r  = t >> 3, kq = t & 7;
                unsigned* dp = &AsU[r * GSTR + kq * 2];
                if (ABF16) {
                    const uint2 a = *(const uint2*)(Abf +
                        (((size_t)(blockRow + r) * (size_t)lda + (size_t)(k0 + kq * 4)) >> 1));
                    dp[0] = a.x; dp[1] = a.y;
                } else {
                    const float* ap = Af + (size_t)(blockRow + r) * lda + k0 + kq * 4;
                    const float4 a = *(const float4*)ap;
                    dp[0] = pack_bf2(a.x, a.y);
                    dp[1] = pack_bf2(a.z, a.w);
                    // prefetch next K-tile of this row (global_prefetch_b8)
                    if (k0 + 32 < K) __builtin_prefetch(ap + 32, 0, 1);
                }
            }
            // B tile: 16 k-pairs x 32 col-quads, transposed into Bs[col][kp]
            #pragma unroll
            for (int it = 0; it < 2; ++it) {
                int t   = tid + it * 256;          // 0..511
                int kp  = t >> 5, c4 = (t & 31) * 4;
                int k   = k0 + kp * 2;
                const float4 b0 = *(const float4*)(B + (size_t)k       * 128 + c4);
                const float4 b1 = *(const float4*)(B + (size_t)(k + 1) * 128 + c4);
                BsU[(c4 + 0) * GSTR + kp] = pack_bf2(b0.x, b1.x);
                BsU[(c4 + 1) * GSTR + kp] = pack_bf2(b0.y, b1.y);
                BsU[(c4 + 2) * GSTR + kp] = pack_bf2(b0.z, b1.z);
                BsU[(c4 + 3) * GSTR + kp] = pack_bf2(b0.w, b1.w);
            }
        } else {
            // ---------------- guarded path (tail M-block / K%32!=0) -------
            for (int t = tid; t < 128 * 16; t += 256) {
                int r  = t >> 4, kp = t & 15;
                int gr = blockRow + r;
                int k  = k0 + kp * 2;
                unsigned val;
                if (ABF16) {
                    val = (gr < M && k < K)
                            ? Abf[((size_t)gr * (size_t)lda + (size_t)k) >> 1] : 0u;
                } else {
                    float a0 = (gr < M && k     < K) ? Af[(size_t)gr * lda + k]     : 0.f;
                    float a1 = (gr < M && k + 1 < K) ? Af[(size_t)gr * lda + k + 1] : 0.f;
                    val = pack_bf2(a0, a1);
                }
                AsU[r * GSTR + kp] = val;
            }
            for (int t = tid; t < 128 * 16; t += 256) {
                int col = t & 127, kp = t >> 7;
                int k   = k0 + kp * 2;
                float b0 = (k     < K) ? B[(size_t)k       * 128 + col] : 0.f;
                float b1 = (k + 1 < K) ? B[(size_t)(k + 1) * 128 + col] : 0.f;
                BsU[col * GSTR + kp] = pack_bf2(b0, b1);
            }
        }
        __syncthreads();

        // ---- build B fragments: lanes 0-15 K={0..7,16..23}, lanes 16-31 K={8..15,24..31}
        FragU fb[4];
        const int colL = (wn << 6) + (lane & 15);
        const int khB  = (lane >> 4) << 2;   // pair offset 0 or 4
        #pragma unroll
        for (int ni = 0; ni < 4; ++ni) {
            const unsigned* bp = &BsU[(colL + ni * 16) * GSTR + khB];
            #pragma unroll
            for (int v = 0; v < 8; ++v) fb[ni].u[v] = bp[(v & 3) + ((v >> 2) << 3)];
        }
        // ---- A fragments: lanes 0-15 K={0..7,16..23}, lanes 16-31 K={8..15,24..31}
        const int khA = (lane >> 4) << 2;    // pair offset 0 or 4
        #pragma unroll
        for (int mi = 0; mi < 2; ++mi) {
            FragU fa;
            const int r = (wm << 5) + (mi << 4) + (lane & 15);
            const unsigned* ap = &AsU[r * GSTR];
            #pragma unroll
            for (int v = 0; v < 4; ++v) fa.u[v]     = ap[khA + v];
            #pragma unroll
            for (int v = 0; v < 4; ++v) fa.u[4 + v] = ap[8 + khA + v];
            #pragma unroll
            for (int ni = 0; ni < 4; ++ni) {
                acc[mi][ni] = __builtin_amdgcn_wmma_f32_16x16x32_bf16(
                    false, fa.v, false, fb[ni].v, (short)0, acc[mi][ni],
                    false, false);
            }
        }
        __syncthreads();
    }

    // ---- epilogue: C/D layout: elem i -> row (lane<16 ? i : 8+i), col lane&15
    const int rowHalf = (lane < 16) ? 0 : 8;
    const int colBase = (wn << 6) + (lane & 15);
    #pragma unroll
    for (int mi = 0; mi < 2; ++mi) {
        #pragma unroll
        for (int ni = 0; ni < 4; ++ni) {
            const int gc = colBase + ni * 16;
            const float bv = bias ? bias[gc] : 0.f;
            #pragma unroll
            for (int i = 0; i < 8; ++i) {
                const int gr = blockRow + (wm << 5) + (mi << 4) + rowHalf + i;
                if (gr < M) {
                    float v = acc[mi][ni][i] + bv;
                    if (doRelu) v = fmaxf(v, 0.f);
                    C[(size_t)gr * 128 + gc] = v;
                }
            }
        }
    }
}

// ---------------------------------------------------------------------------
// Small helper kernels
// ---------------------------------------------------------------------------
__global__ void fill_f32(float* p, float v, int n) {
    int i = blockIdx.x * blockDim.x + threadIdx.x;
    if (i < n) p[i] = v;
}

__global__ void init_acc(float* s, float* q, unsigned* mn, unsigned* mx, int n) {
    int i = blockIdx.x * blockDim.x + threadIdx.x;
    if (i < n) { s[i] = 0.f; q[i] = 0.f; mn[i] = 0xFFFFFFFFu; mx[i] = 0u; }
}

__global__ void deg_kernel(const int* __restrict__ dst, float* deg, int E) {
    int i = blockIdx.x * blockDim.x + threadIdx.x;
    if (i < E) atomicAdd(&deg[dst[i]], 1.0f);
}

__global__ void avglog_kernel(const float* __restrict__ deg, float* out, int n) {
    __shared__ float red[256];
    int i = blockIdx.x * 256 + threadIdx.x;
    red[threadIdx.x] = (i < n) ? log1pf(deg[i]) : 0.f;
    __syncthreads();
    for (int s = 128; s > 0; s >>= 1) {
        if (threadIdx.x < s) red[threadIdx.x] += red[threadIdx.x + s];
        __syncthreads();
    }
    if (threadIdx.x == 0) atomicAdd(out, red[0]);
}

// combined weights: CW[l] = encW[l] @ preW3[l] ; cb[l] = encB[l]@preW3[l]+preB[l]
__global__ void cw_kernel(const float* __restrict__ encW, const float* __restrict__ encB,
                          const float* __restrict__ preW, const float* __restrict__ preB,
                          float* CW, float* cb, int L)
{
    int idx = blockIdx.x * blockDim.x + threadIdx.x;
    int tot = L * 129 * 128;
    if (idx >= tot) return;
    int l = idx / (129 * 128);
    int rem = idx - l * 129 * 128;
    int r = rem >> 7, j = rem & 127;
    const float* W3 = preW + (size_t)l * 384 * 128 + (size_t)256 * 128;  // rows 2H..3H
    float s = 0.f;
    if (r < 128) {
        const float* a = encW + (size_t)l * 128 * 128 + (size_t)r * 128;
        for (int i = 0; i < 128; ++i) s += a[i] * W3[(size_t)i * 128 + j];
        CW[(size_t)l * 128 * 128 + (size_t)r * 128 + j] = s;
    } else {
        const float* b = encB + (size_t)l * 128;
        for (int i = 0; i < 128; ++i) s += b[i] * W3[(size_t)i * 128 + j];
        cb[(size_t)l * 128 + j] = s + preB[(size_t)l * 128 + j];
    }
}

// per-edge message + scatter-aggregate (sum, sumsq, min, max) into L2-resident accs
__global__ void edge_agg(const float* __restrict__ hp1, const float* __restrict__ hp2,
                         const float* __restrict__ eW3,
                         const int* __restrict__ src, const int* __restrict__ dst,
                         float* accS, float* accQ, unsigned* accMn, unsigned* accMx, int E)
{
    int idx = blockIdx.x * blockDim.x + threadIdx.x;   // 32 threads/edge, 4 feats each
    int e = idx >> 5;
    if (e >= E) return;
    int f4 = (idx & 31) << 2;
    int d = dst[e], s = src[e];
    float4 a = *(const float4*)(hp1 + (size_t)d * 128 + f4);
    float4 b = *(const float4*)(hp2 + (size_t)s * 128 + f4);
    float4 c = *(const float4*)(eW3 + (size_t)e * 128 + f4);
    float m[4] = { a.x + b.x + c.x, a.y + b.y + c.y,
                   a.z + b.z + c.z, a.w + b.w + c.w };
    size_t base = (size_t)d * 128 + f4;
    #pragma unroll
    for (int i = 0; i < 4; ++i) {
        atomicAdd(&accS[base + i], m[i]);
        atomicAdd(&accQ[base + i], m[i] * m[i]);
        atomicMin(&accMn[base + i], fenc(m[i]));
        atomicMax(&accMx[base + i], fenc(m[i]));
    }
}

__device__ __forceinline__ void stbf4(unsigned short* dstp, float a0, float a1,
                                      float a2, float a3) {
    unsigned* d2 = (unsigned*)dstp;
    d2[0] = pack_bf2(a0, a1);
    d2[1] = pack_bf2(a2, a3);
}

// build X13[n, 13*128] (bf16) = [h | mean mn mx std | *amp | *att]
__global__ void finalize_agg(const float* __restrict__ h,
                             const float* __restrict__ accS, const float* __restrict__ accQ,
                             const unsigned* __restrict__ accMn, const unsigned* __restrict__ accMx,
                             const float* __restrict__ deg, const float* __restrict__ avgSum,
                             unsigned short* __restrict__ X13, int Nn)
{
    int idx = blockIdx.x * blockDim.x + threadIdx.x;   // 32 threads/node, 4 feats each
    int n = idx >> 5;
    if (n >= Nn) return;
    int f4 = (idx & 31) << 2;

    float dv   = deg[n];
    float degc = fmaxf(dv, 1.0f);
    float al   = avgSum[0] / (float)Nn;
    float la   = logf(degc + 1.0f);
    float amp  = la / al;
    float att  = al / la;
    bool  has  = dv > 0.f;

    size_t base = (size_t)n * 128 + f4;
    float4 s4 = *(const float4*)(accS + base);
    float4 q4 = *(const float4*)(accQ + base);
    float mean[4], mn[4], mx[4], st[4];
    #pragma unroll
    for (int i = 0; i < 4; ++i) {
        float s = ((const float*)&s4)[i];
        float q = ((const float*)&q4)[i];
        mean[i] = s / degc;
        float var = q / degc - mean[i] * mean[i];
        st[i] = sqrtf(fmaxf(var, 0.f) + 1e-5f);
        mn[i] = has ? fdec(accMn[base + i]) : 0.f;
        mx[i] = has ? fdec(accMx[base + i]) : 0.f;
    }
    float4 h4 = *(const float4*)(h + base);

    unsigned short* row = X13 + (size_t)n * 1664 + f4;
    stbf4(row + 0 * 128, h4.x, h4.y, h4.z, h4.w);
    // agg
    stbf4(row + 1 * 128, mean[0], mean[1], mean[2], mean[3]);
    stbf4(row + 2 * 128, mn[0], mn[1], mn[2], mn[3]);
    stbf4(row + 3 * 128, mx[0], mx[1], mx[2], mx[3]);
    stbf4(row + 4 * 128, st[0], st[1], st[2], st[3]);
    // agg * amp
    stbf4(row + 5 * 128, mean[0]*amp, mean[1]*amp, mean[2]*amp, mean[3]*amp);
    stbf4(row + 6 * 128, mn[0]*amp,   mn[1]*amp,   mn[2]*amp,   mn[3]*amp);
    stbf4(row + 7 * 128, mx[0]*amp,   mx[1]*amp,   mx[2]*amp,   mx[3]*amp);
    stbf4(row + 8 * 128, st[0]*amp,   st[1]*amp,   st[2]*amp,   st[3]*amp);
    // agg * att
    stbf4(row + 9 * 128, mean[0]*att, mean[1]*att, mean[2]*att, mean[3]*att);
    stbf4(row +10 * 128, mn[0]*att,   mn[1]*att,   mn[2]*att,   mn[3]*att);
    stbf4(row +11 * 128, mx[0]*att,   mx[1]*att,   mx[2]*att,   mx[3]*att);
    stbf4(row +12 * 128, st[0]*att,   st[1]*att,   st[2]*att,   st[3]*att);
}

__global__ void pool_kernel(const float* __restrict__ h, const int* __restrict__ batch,
                            float* pooled, int Nn)
{
    int idx = blockIdx.x * blockDim.x + threadIdx.x;   // 32 threads/node, 4 feats
    int n = idx >> 5;
    if (n >= Nn) return;
    int f4 = (idx & 31) << 2;
    int g = batch[n];
    float4 v = *(const float4*)(h + (size_t)n * 128 + f4);
    atomicAdd(&pooled[(size_t)g * 128 + f4 + 0], v.x);
    atomicAdd(&pooled[(size_t)g * 128 + f4 + 1], v.y);
    atomicAdd(&pooled[(size_t)g * 128 + f4 + 2], v.z);
    atomicAdd(&pooled[(size_t)g * 128 + f4 + 3], v.w);
}

__global__ void head_kernel(const float* __restrict__ pooled,
                            const float* __restrict__ hW, const float* __restrict__ hb,
                            float* out, int G, int OUT)
{
    int idx = blockIdx.x * blockDim.x + threadIdx.x;
    if (idx >= G * OUT) return;
    int g = idx / OUT, o = idx - g * OUT;
    float s = hb[o];
    const float* p = pooled + (size_t)g * 128;
    for (int k = 0; k < 128; ++k) s += p[k] * hW[(size_t)k * OUT + o];
    out[idx] = s;
}

// ---------------------------------------------------------------------------
// Host launcher
// ---------------------------------------------------------------------------
static inline void launch_gemm(const void* A, int aIsBf16, int lda, const float* B,
                               const float* bias, float* C, int M, int K, int relu,
                               hipStream_t s)
{
    dim3 grid((M + 127) / 128);
    if (aIsBf16)
        gemm_wmma<1><<<grid, 256, 0, s>>>(A, lda, B, bias, C, M, K, relu);
    else
        gemm_wmma<0><<<grid, 256, 0, s>>>(A, lda, B, bias, C, M, K, relu);
}

extern "C" void kernel_launch(void* const* d_in, const int* in_sizes, int n_in,
                              void* d_out, int out_size, void* d_ws, size_t ws_size,
                              hipStream_t stream)
{
    const float* x          = (const float*)d_in[0];
    const float* edge_attr  = (const float*)d_in[1];
    const int*   src        = (const int*)d_in[2];
    const int*   dst        = (const int*)d_in[3];
    const int*   batch      = (const int*)d_in[4];
    const float* node_emb_W = (const float*)d_in[5];
    const float* node_emb_b = (const float*)d_in[6];
    const float* edge_emb_W = (const float*)d_in[7];
    const float* edge_emb_b = (const float*)d_in[8];
    const float* edge_enc_W = (const float*)d_in[9];
    const float* edge_enc_b = (const float*)d_in[10];
    const float* pre_W      = (const float*)d_in[11];
    const float* pre_b      = (const float*)d_in[12];
    const float* post_W     = (const float*)d_in[13];
    const float* post_b     = (const float*)d_in[14];
    const float* lin_W      = (const float*)d_in[15];
    const float* lin_b      = (const float*)d_in[16];
    const float* head_W     = (const float*)d_in[17];
    const float* head_b     = (const float*)d_in[18];

    const int N    = in_sizes[4];              // batch vector length
    const int E    = in_sizes[2];              // src length
    const int F_IN = in_sizes[0] / N;
    const int E_IN = in_sizes[1] / E;
    const int L    = in_sizes[10] / 128;       // edge_enc_b = [L,128]
    const int OUT  = in_sizes[18];
    const int G    = out_size / OUT;

    // ---- carve workspace ----
    char* wp = (char*)d_ws;
    auto carve = [&](size_t bytes) -> void* {
        void* r = (void*)wp;
        wp += (bytes + 255) & ~(size_t)255;
        return r;
    };
    float*          hA     = (float*)carve((size_t)N * 128 * 4);
    float*          hB     = (float*)carve((size_t)N * 128 * 4);
    float*          outB   = (float*)carve((size_t)N * 128 * 4);
    float*          e0     = (float*)carve((size_t)E * 128 * 4);
    float*          eW3    = (float*)carve((size_t)E * 128 * 4);
    float*          hp1    = (float*)carve((size_t)N * 128 * 4);
    float*          hp2    = (float*)carve((size_t)N * 128 * 4);
    float*          accS   = (float*)carve((size_t)N * 128 * 4);
    float*          accQ   = (float*)carve((size_t)N * 128 * 4);
    unsigned*       accMn  = (unsigned*)carve((size_t)N * 128 * 4);
    unsigned*       accMx  = (unsigned*)carve((size_t)N * 128 * 4);
    unsigned short* X13    = (unsigned short*)carve((size_t)N * 1664 * 2);
    float*          deg    = (float*)carve((size_t)N * 4);
    float*          avgSum = (float*)carve(256);
    float*          pooled = (float*)carve((size_t)G * 128 * 4);
    float*          CW     = (float*)carve((size_t)L * 128 * 128 * 4);
    float*          cb     = (float*)carve((size_t)L * 128 * 4);
    (void)ws_size; (void)n_in;

    const int TB = 256;
    auto blocks = [&](long long n) { return (unsigned)((n + TB - 1) / TB); };

    // ---- prologue: degrees, avg log degree, combined weights, embeddings ----
    fill_f32<<<blocks(N), TB, 0, stream>>>(deg, 0.f, N);
    fill_f32<<<1, TB, 0, stream>>>(avgSum, 0.f, 1);
    fill_f32<<<blocks((long long)G * 128), TB, 0, stream>>>(pooled, 0.f, G * 128);
    cw_kernel<<<blocks((long long)L * 129 * 128), TB, 0, stream>>>(
        edge_enc_W, edge_enc_b, pre_W, pre_b, CW, cb, L);

    launch_gemm(x,         0, F_IN, node_emb_W, node_emb_b, hA, N, F_IN, 0, stream);
    launch_gemm(edge_attr, 0, E_IN, edge_emb_W, edge_emb_b, e0, E, E_IN, 0, stream);

    deg_kernel<<<blocks(E), TB, 0, stream>>>(dst, deg, E);
    avglog_kernel<<<blocks(N), TB, 0, stream>>>(deg, avgSum, N);

    // ---- layers ----
    for (int l = 0; l < L; ++l) {
        init_acc<<<blocks((long long)N * 128), TB, 0, stream>>>(
            accS, accQ, accMn, accMx, N * 128);

        const float* W12 = pre_W + (size_t)l * 384 * 128;       // rows [0,2H)
        launch_gemm(hA, 0, 128, W12,             nullptr, hp1, N, 128, 0, stream);
        launch_gemm(hA, 0, 128, W12 + 128 * 128, nullptr, hp2, N, 128, 0, stream);
        launch_gemm(e0, 0, 128, CW + (size_t)l * 128 * 128, cb + (size_t)l * 128,
                    eW3, E, 128, 0, stream);

        edge_agg<<<blocks((long long)E * 32), TB, 0, stream>>>(
            hp1, hp2, eW3, src, dst, accS, accQ, accMn, accMx, E);

        finalize_agg<<<blocks((long long)N * 32), TB, 0, stream>>>(
            hA, accS, accQ, accMn, accMx, deg, avgSum, X13, N);

        launch_gemm(X13, 1, 1664, post_W + (size_t)l * 1664 * 128,
                    post_b + (size_t)l * 128, outB, N, 1664, 0, stream);
        launch_gemm(outB, 0, 128, lin_W + (size_t)l * 128 * 128,
                    lin_b + (size_t)l * 128, hB, N, 128, 1, stream);

        float* t = hA; hA = hB; hB = t;
    }

    // ---- pooling + head ----
    pool_kernel<<<blocks((long long)N * 32), TB, 0, stream>>>(hA, batch, pooled, N);
    head_kernel<<<blocks((long long)G * OUT), TB, 0, stream>>>(
        pooled, head_W, head_b, (float*)d_out, G, OUT);
}
